// GCN_10539849744470
// MI455X (gfx1250) — matrix operand order repaired and checked by reading, not compiled, not run on hardware
//
#include <hip/hip_runtime.h>

#define NN 20000
#define NE 320000
#define HID 256
#define NCLS 5

typedef __attribute__((ext_vector_type(16))) __bf16 v16bf;
typedef __attribute__((ext_vector_type(8)))  __bf16 v8bf;
typedef __attribute__((ext_vector_type(8)))  float  v8f;

// ---------- helpers ----------
__device__ __forceinline__ __bf16 f2bf(float f) {
    unsigned u = __builtin_bit_cast(unsigned, f);
    u += 0x7FFFu + ((u >> 16) & 1u);            // round-to-nearest-even
    unsigned short h = (unsigned short)(u >> 16);
    return __builtin_bit_cast(__bf16, h);
}

// A fragment (16x32 bf16): lane = M row; lanes 0-15 hold K=kk..kk+15, lanes 16-31 hold K=kk+16..kk+31
__device__ __forceinline__ v16bf load_a_frag(const __bf16* __restrict__ row, int kk, int half) {
    return *(const v16bf*)(row + kk + 16 * half);
}

// B fragment (32x16 bf16): lane = N col; lanes 0-15 hold K=kk..kk+15, lanes 16-31 hold K=kk+16..kk+31
__device__ __forceinline__ v16bf load_b_frag(const __bf16* __restrict__ row, int kk, int half) {
    return *(const v16bf*)(row + kk + 16 * half);   // 32 contiguous bytes
}

// ---------- kernels ----------

// Transpose + convert one 256x256 fp32 weight to bf16 WT[n][k] = W[k][n]
__global__ void wt_convert(const float* __restrict__ W, __bf16* __restrict__ WT) {
    int n = blockIdx.x, k = threadIdx.x;
    WT[n * HID + k] = f2bf(W[k * HID + n]);
}

__global__ void zero256(float* __restrict__ p) { p[threadIdx.x] = 0.0f; }

__global__ void deg_init(float* __restrict__ deg, int n) {
    int i = blockIdx.x * blockDim.x + threadIdx.x;
    if (i < n) deg[i] = 1.0f;                       // self-loop
}

__global__ void deg_edges(const int* __restrict__ dst, float* __restrict__ deg, int e) {
    int i = blockIdx.x * blockDim.x + threadIdx.x;
    if (i < e) atomicAdd(&deg[dst[i]], 1.0f);
}

__global__ void deg_finish(float* __restrict__ deg, int n) {
    int i = blockIdx.x * blockDim.x + threadIdx.x;
    if (i < n) deg[i] = rsqrtf(deg[i]);             // in-place -> dinv
}

__global__ void cvt_relu(const float* __restrict__ in, __bf16* __restrict__ out,
                         int n, int do_relu) {
    int i = blockIdx.x * blockDim.x + threadIdx.x;
    if (i < n) {
        float v = in[i];
        if (do_relu) v = fmaxf(v, 0.0f);
        out[i] = f2bf(v);
    }
}

// C[M x 256] = A[M x 256] (bf16) @ W (via WT bf16 [256n][256k]); fp32 accumulate.
// Block = 256 threads (8 waves); block tile = 32(M) x 256(N); wave tile = 32x32 (4 WMMAs/K-step).
__global__ void __launch_bounds__(256)
gemm_bf16_wmma(const __bf16* __restrict__ A, const __bf16* __restrict__ WT,
               float* __restrict__ C) {
    const int tid  = threadIdx.x;
    const int wave = tid >> 5;
    const int lane = tid & 31;
    const int half = lane >> 4;
    const int l16  = lane & 15;
    const int mBase = blockIdx.x * 32;
    const int nBase = wave * 32;

    const __bf16* a0 = A  + (size_t)(mBase + l16) * HID;
    const __bf16* a1 = a0 + (size_t)16 * HID;
    const __bf16* b0 = WT + (size_t)(nBase + l16) * HID;
    const __bf16* b1 = b0 + (size_t)16 * HID;

    v8f acc[2][2] = {};
#pragma unroll
    for (int kk = 0; kk < HID; kk += 32) {
        v16bf fa0 = load_a_frag(a0, kk, half);
        v16bf fa1 = load_a_frag(a1, kk, half);
        v16bf fb0 = load_b_frag(b0, kk, half);
        v16bf fb1 = load_b_frag(b1, kk, half);
        acc[0][0] = __builtin_amdgcn_wmma_f32_16x16x32_bf16(false, fa0, false, fb0, (short)0, acc[0][0], false, false);
        acc[0][1] = __builtin_amdgcn_wmma_f32_16x16x32_bf16(false, fa0, false, fb1, (short)0, acc[0][1], false, false);
        acc[1][0] = __builtin_amdgcn_wmma_f32_16x16x32_bf16(false, fa1, false, fb0, (short)0, acc[1][0], false, false);
        acc[1][1] = __builtin_amdgcn_wmma_f32_16x16x32_bf16(false, fa1, false, fb1, (short)0, acc[1][1], false, false);
    }
    // C/D layout: lane 0-15: N=lane, M=r; lanes 16-31: N=lane-16, M=r+8
#pragma unroll
    for (int i = 0; i < 2; ++i)
#pragma unroll
        for (int j = 0; j < 2; ++j)
#pragma unroll
            for (int r = 0; r < 8; ++r) {
                int row = mBase + i * 16 + r + 8 * half;
                int col = nBase + j * 16 + l16;
                C[(size_t)row * HID + col] = acc[i][j][r];
            }
}

// agg[i][f] = xw[i][f] * dinv[i]^2 + b[f]   (self-loop term + bias init)
__global__ void self_term(const float* __restrict__ xw, const float* __restrict__ dinv,
                          const float* __restrict__ b, float* __restrict__ agg, int n) {
    int idx = blockIdx.x * blockDim.x + threadIdx.x;
    if (idx < n * HID) {
        int row = idx >> 8, f = idx & 255;
        float di = dinv[row];
        agg[idx] = xw[idx] * di * di + b[f];
    }
}

// One wave per edge: agg[dst] += xw[src] * dinv[src]*dinv[dst]; lane covers 8 features (2x float4).
__global__ void __launch_bounds__(256)
edge_agg(const int* __restrict__ src, const int* __restrict__ dst,
         const float* __restrict__ dinv, const float* __restrict__ xw,
         float* __restrict__ agg, int e) {
    int edge = blockIdx.x * 8 + (threadIdx.x >> 5);
    if (edge >= e) return;
    int lane = threadIdx.x & 31;
    int s = src[edge], d = dst[edge];
    float coef = dinv[s] * dinv[d];
    const float4* xr = (const float4*)(xw + (size_t)s * HID);
    float* ar = agg + (size_t)d * HID;
    float4 v0 = xr[lane * 2 + 0];
    float4 v1 = xr[lane * 2 + 1];
    int f = lane * 8;
    atomicAdd(ar + f + 0, v0.x * coef);
    atomicAdd(ar + f + 1, v0.y * coef);
    atomicAdd(ar + f + 2, v0.z * coef);
    atomicAdd(ar + f + 3, v0.w * coef);
    atomicAdd(ar + f + 4, v1.x * coef);
    atomicAdd(ar + f + 5, v1.y * coef);
    atomicAdd(ar + f + 6, v1.z * coef);
    atomicAdd(ar + f + 7, v1.w * coef);
}

// colsum[f] += sum over 64 rows of h[r][f]
__global__ void colsum_reduce(const float* __restrict__ h, float* __restrict__ colsum, int n) {
    int f = threadIdx.x;
    int r0 = blockIdx.x * 64;
    int r1 = (r0 + 64 < n) ? (r0 + 64) : n;
    float s = 0.0f;
    for (int r = r0; r < r1; ++r) s += h[(size_t)r * HID + f];
    atomicAdd(&colsum[f], s);
}

// out[c] = (colsum/2N) @ Wl + bl
__global__ void final_linear(const float* __restrict__ colsum, const float* __restrict__ Wl,
                             const float* __restrict__ bl, float* __restrict__ out) {
    int c = threadIdx.x;
    if (c < NCLS) {
        float s = 0.0f;
        for (int k = 0; k < HID; ++k) s += colsum[k] * Wl[k * NCLS + c];
        out[c] = s * (1.0f / (2.0f * NN)) + bl[c];
    }
}

// ---------- launch ----------
extern "C" void kernel_launch(void* const* d_in, const int* in_sizes, int n_in,
                              void* d_out, int out_size, void* d_ws, size_t ws_size,
                              hipStream_t stream) {
    const float* x [2] = {(const float*)d_in[0], (const float*)d_in[2]};
    const int*   ei[2] = {(const int*)  d_in[1], (const int*)  d_in[3]};
    const float* W [3] = {(const float*)d_in[4], (const float*)d_in[6], (const float*)d_in[8]};
    const float* b [3] = {(const float*)d_in[5], (const float*)d_in[7], (const float*)d_in[9]};
    const float* Wl = (const float*)d_in[10];
    const float* bl = (const float*)d_in[11];

    char* p = (char*)d_ws;                                   // ~52 MB total, 16B-aligned carve
    __bf16* WT   = (__bf16*)p; p += (size_t)3 * HID * HID * sizeof(__bf16);
    float*  dinv = (float*) p; p += (size_t)NN * sizeof(float);
    __bf16* Xb   = (__bf16*)p; p += (size_t)NN * HID * sizeof(__bf16);
    float*  XW   = (float*) p; p += (size_t)NN * HID * sizeof(float);
    float*  AGG  = (float*) p; p += (size_t)NN * HID * sizeof(float);
    float*  csum = (float*) p; p += (size_t)HID * sizeof(float);

    for (int l = 0; l < 3; ++l)
        wt_convert<<<HID, HID, 0, stream>>>(W[l], WT + (size_t)l * HID * HID);
    zero256<<<1, HID, 0, stream>>>(csum);

    for (int br = 0; br < 2; ++br) {
        deg_init  <<<(NN + 255) / 256, 256, 0, stream>>>(dinv, NN);
        deg_edges <<<(NE + 255) / 256, 256, 0, stream>>>(ei[br] + NE, dinv, NE);
        deg_finish<<<(NN + 255) / 256, 256, 0, stream>>>(dinv, NN);
        cvt_relu  <<<NN, 256, 0, stream>>>(x[br], Xb, NN * HID, 0);
        for (int l = 0; l < 3; ++l) {
            gemm_bf16_wmma<<<NN / 32, 256, 0, stream>>>(Xb, WT + (size_t)l * HID * HID, XW);
            self_term<<<NN, 256, 0, stream>>>(XW, dinv, b[l], AGG, NN);
            edge_agg<<<NE / 8, 256, 0, stream>>>(ei[br], ei[br] + NE, dinv, XW, AGG, NE);
            if (l < 2)
                cvt_relu<<<NN, 256, 0, stream>>>(AGG, Xb, NN * HID, 1);
        }
        colsum_reduce<<<(NN + 63) / 64, 256, 0, stream>>>(AGG, csum, NN);
    }
    final_linear<<<1, 32, 0, stream>>>(csum, Wl, bl, (float*)d_out);
}
